// Self_Attention_51333449122099
// MI455X (gfx1250) — matrix operand order repaired and checked by reading, not compiled, not run on hardware
//
#include <hip/hip_runtime.h>

// Problem constants (B,H,S,D from the reference)
#define BB 2
#define HH 16
#define SS 2048
#define DD 64
#define BHN (BB * HH)
#define QTILE 128   // query rows per block (8 waves x 16 rows)
#define KT 64       // keys per inner iteration
#define NWAVE 8

typedef __attribute__((ext_vector_type(16))) __bf16 v16bf;
typedef __attribute__((ext_vector_type(4)))  __bf16 v4bf;
typedef __attribute__((ext_vector_type(8)))  float  v8f;
typedef __attribute__((ext_vector_type(4)))  unsigned int u32x4;
typedef __attribute__((ext_vector_type(8)))  int  i32x8;
typedef __attribute__((ext_vector_type(4)))  int  i32x4;

union U16 {
    uint4  u[2];
    v16bf  v;
};

#if __has_builtin(__builtin_amdgcn_tensor_load_to_lds)
#define HAVE_TDM 1
#else
#define HAVE_TDM 0
#endif

#define LOG2E 1.44269504088896340736f

// ---------------------------------------------------------------------------
// Pass 1a: fp32 -> bf16 conversion (Q, K), row-major, bandwidth bound
// ---------------------------------------------------------------------------
__global__ __launch_bounds__(256) void cvt_f32_to_bf16(const float* __restrict__ src,
                                                       __bf16* __restrict__ dst,
                                                       unsigned n) {
    unsigned i = (blockIdx.x * 256u + threadIdx.x) * 4u;
    if (i < n) {
        float4 f = *(const float4*)(src + i);
        v4bf o;
        o[0] = (__bf16)f.x;
        o[1] = (__bf16)f.y;
        o[2] = (__bf16)f.z;
        o[3] = (__bf16)f.w;
        *(v4bf*)(dst + i) = o;
    }
}

// ---------------------------------------------------------------------------
// Pass 1b: V fp32 [bh][s][d] -> bf16 [bh][d][s] (transpose once, not per block)
// One 64x64 tile per 256-thread block via padded LDS.
// ---------------------------------------------------------------------------
__global__ __launch_bounds__(256) void cvt_transpose_v(const float* __restrict__ src,
                                                       __bf16* __restrict__ dst) {
    __shared__ __align__(16) __bf16 t[DD][DD + 8];   // +8 pad kills bank conflicts
    const int bh = blockIdx.y;
    const int s0 = blockIdx.x * 64;

    // Read 64x64 f32 tile coalesced: thread -> (key row, 16-wide d chunk)
    const int key = threadIdx.x >> 2;
    const int c   = (threadIdx.x & 3) * 16;
    const float* srow = src + ((size_t)bh * SS + s0 + key) * DD + c;
    float4 f0 = ((const float4*)srow)[0];
    float4 f1 = ((const float4*)srow)[1];
    float4 f2 = ((const float4*)srow)[2];
    float4 f3 = ((const float4*)srow)[3];
    float ff[16] = {f0.x, f0.y, f0.z, f0.w, f1.x, f1.y, f1.z, f1.w,
                    f2.x, f2.y, f2.z, f2.w, f3.x, f3.y, f3.z, f3.w};
#pragma unroll
    for (int i = 0; i < 16; ++i) t[c + i][key] = (__bf16)ff[i];
    __syncthreads();

    // Write V^T coalesced: thread -> (d row, 16-wide key chunk)
    const int d  = threadIdx.x >> 2;
    const int kc = (threadIdx.x & 3) * 16;
    __bf16* drow = dst + ((size_t)bh * DD + d) * SS + s0 + kc;
    *(uint4*)(drow)     = *(const uint4*)&t[d][kc];
    *(uint4*)(drow + 8) = *(const uint4*)&t[d][kc + 8];
}

#if HAVE_TDM
// Get the 32-bit LDS byte offset of a shared-memory object.
__device__ __forceinline__ unsigned lds_off_u32(const void* p) {
    typedef __attribute__((address_space(3))) const void as3_void;
    return (unsigned)(unsigned long long)(as3_void*)p;
}

// TDM: DMA a 64-row x 64-elem bf16 tile (row stride `stride` elems) into LDS.
__device__ __forceinline__ void tdm_load_tile(unsigned lds_byte_off, const __bf16* gptr,
                                              unsigned stride) {
    const unsigned long long ga = (unsigned long long)(size_t)gptr;
    u32x4 g0;
    g0[0] = 1u;                                          // count=1, user descriptor
    g0[1] = lds_byte_off;                                // lds_addr
    g0[2] = (unsigned)(ga & 0xFFFFFFFFu);                // global_addr[31:0]
    g0[3] = (unsigned)((ga >> 32) & 0x01FFFFFFu)         // global_addr[56:32]
          | (2u << 30);                                  // type=2 ("image")
    i32x8 g1;
    g1[0] = 0x10000;                                     // data_size=1 (2 bytes)
    g1[1] = (int)((stride & 0xFFFFu) << 16);             // tensor_dim0 = stride
    g1[2] = (int)((stride >> 16) | ((unsigned)SS << 16));// tensor_dim1 = 2048
    g1[3] = (int)(64u << 16);                            // tile_dim0 = 64
    g1[4] = 64;                                          // tile_dim1 = 64 rows
    g1[5] = (int)stride;                                 // tensor_dim0_stride
    g1[6] = 0;
    g1[7] = 0;
    i32x4 z = {0, 0, 0, 0};
#if defined(__clang_major__) && (__clang_major__ >= 23)
    i32x8 z8 = {0, 0, 0, 0, 0, 0, 0, 0};
    __builtin_amdgcn_tensor_load_to_lds(g0, g1, z, z, z8, 0);
#else
    __builtin_amdgcn_tensor_load_to_lds(g0, g1, z, z, 0);
#endif
}
#endif  // HAVE_TDM

#define WMMA_BF16(A, B, C) \
    __builtin_amdgcn_wmma_f32_16x16x32_bf16(false, (A), false, (B), (short)0, (C), false, false)

// ---------------------------------------------------------------------------
// Pass 2: flash attention, bf16 WMMA, TDM double-buffered K/V^T tiles
// ---------------------------------------------------------------------------
__global__ __launch_bounds__(256) void fa_fwd_kernel(const __bf16* __restrict__ qg,
                                                     const __bf16* __restrict__ kg,
                                                     const __bf16* __restrict__ vtg,
                                                     const int* __restrict__ is_masked_p,
                                                     const int* __restrict__ padding_p,
                                                     float* __restrict__ out) {
    __shared__ __align__(16) __bf16 lds_k[2][KT][DD];     // K tile, row major (key, d)
    __shared__ __align__(16) __bf16 lds_v[2][DD][KT];     // V^T tile (d, key)
    __shared__ __align__(16) __bf16 lds_p[NWAVE][16][KT]; // per-wave P staging (row, key)

    const int tid   = threadIdx.x;
    const int wave  = tid >> 5;
    const int lane  = tid & 31;
    const int lhalf = lane >> 4;   // 0: lanes 0-15, 1: lanes 16-31
    const int l16   = lane & 15;

    const int bh    = blockIdx.y;              // fused (b,h)
    const int qb0   = blockIdx.x * QTILE;      // first query row of this block
    const int qrow0 = qb0 + wave * 16;         // first query row of this wave

    const int   masked = *is_masked_p;
    const float padf2  = (float)(*padding_p) * LOG2E;  // padding compare in log2 domain
    const float scale2 = 0.125f * LOG2E;               // 1/sqrt(64) * log2(e)

    const __bf16* Qp  = qg  + (size_t)bh * SS * DD;
    const __bf16* Kp  = kg  + (size_t)bh * SS * DD;
    const __bf16* Vtp = vtg + (size_t)bh * DD * SS;    // [d][s]

    // ---- Load this wave's 16x64 Q tile as two bf16 A-operands (16x32 each).
    U16 qa0, qa1;
    {
        const __bf16* qrow = Qp + (size_t)(qrow0 + l16) * DD + lhalf * 16;
        qa0.u[0] = *(const uint4*)(qrow);
        qa0.u[1] = *(const uint4*)(qrow + 8);
        qa1.u[0] = *(const uint4*)(qrow + 32);
        qa1.u[1] = *(const uint4*)(qrow + 40);
    }

    // All-ones B operand: row-sum of P via WMMA (replaces shuffle reductions).
    v16bf onesb;
#pragma unroll
    for (int i = 0; i < 16; ++i) onesb[i] = (__bf16)1.0f;

    float mrun[8];
#pragma unroll
    for (int r = 0; r < 8; ++r) mrun[r] = -3.0e38f;

    v8f lacc = {};                              // running row sums (ell), O-row layout
    v8f o0 = {}, o1 = {}, o2 = {}, o3 = {};     // O accum: d chunks of 16

    const int jend   = masked ? (qb0 + QTILE) : SS;
    const int ntiles = jend / KT;

    // ---- Prologue: tile 0 into buffer 0
#if HAVE_TDM
    if (wave == 0) {
        tdm_load_tile(lds_off_u32(&lds_k[0][0][0]), Kp, DD);
        tdm_load_tile(lds_off_u32(&lds_v[0][0][0]), Vtp, SS);
        __builtin_amdgcn_s_wait_tensorcnt(0);
    }
#else
    {
        const int row = tid >> 2;
        const int c   = (tid & 3) * 16;
        const __bf16* ks = Kp + (size_t)row * DD + c;
        *(uint4*)&lds_k[0][row][c]     = *(const uint4*)(ks);
        *(uint4*)&lds_k[0][row][c + 8] = *(const uint4*)(ks + 8);
        const __bf16* vs = Vtp + (size_t)row * SS + c;
        *(uint4*)&lds_v[0][row][c]     = *(const uint4*)(vs);
        *(uint4*)&lds_v[0][row][c + 8] = *(const uint4*)(vs + 8);
    }
#endif
    __syncthreads();

    for (int it = 0; it < ntiles; ++it) {
        const int cur = it & 1;
        const int j0  = it * KT;

        // L2 prefetch two tiles ahead (TDM consumes from L2).
        if (j0 + 2 * KT < jend) {
            __builtin_prefetch(Kp + (size_t)(j0 + 2 * KT) * DD + tid * 16, 0, 1);
            __builtin_prefetch(Vtp + (size_t)(tid >> 2) * SS + (j0 + 2 * KT) + (tid & 3) * 16, 0, 1);
        }

#if HAVE_TDM
        // Kick off DMA of the next tile into the other buffer; its previous
        // consumers finished before the barrier we just passed.
        if (wave == 0 && it + 1 < ntiles) {
            tdm_load_tile(lds_off_u32(&lds_k[1 - cur][0][0]), Kp + (size_t)(j0 + KT) * DD, DD);
            tdm_load_tile(lds_off_u32(&lds_v[1 - cur][0][0]), Vtp + (j0 + KT), SS);
        }
#endif

        // Fully-masked tiles (keys all above this wave's rows): skip compute,
        // keep barriers. Uniform branch -> EXEC all-ones for WMMA.
        const bool live = (!masked) || (j0 <= qrow0 + 15);
        if (live) {
            const bool need_mask = masked && (j0 + KT - 1 > qrow0);

            // ---- S = Q*K^T for 16 rows x 64 keys (8 WMMAs, f32 accum)
            v8f acc0 = {}, acc1 = {}, acc2 = {}, acc3 = {};
            const int dlo = lhalf * 16;
            {
                U16 b0, b1;
                b0.u[0] = *(const uint4*)&lds_k[cur][l16][dlo];
                b0.u[1] = *(const uint4*)&lds_k[cur][l16][dlo + 8];
                b1.u[0] = *(const uint4*)&lds_k[cur][l16][32 + dlo];
                b1.u[1] = *(const uint4*)&lds_k[cur][l16][32 + dlo + 8];
                acc0 = WMMA_BF16(qa0.v, b0.v, acc0);
                acc0 = WMMA_BF16(qa1.v, b1.v, acc0);
                b0.u[0] = *(const uint4*)&lds_k[cur][16 + l16][dlo];
                b0.u[1] = *(const uint4*)&lds_k[cur][16 + l16][dlo + 8];
                b1.u[0] = *(const uint4*)&lds_k[cur][16 + l16][32 + dlo];
                b1.u[1] = *(const uint4*)&lds_k[cur][16 + l16][32 + dlo + 8];
                acc1 = WMMA_BF16(qa0.v, b0.v, acc1);
                acc1 = WMMA_BF16(qa1.v, b1.v, acc1);
                b0.u[0] = *(const uint4*)&lds_k[cur][32 + l16][dlo];
                b0.u[1] = *(const uint4*)&lds_k[cur][32 + l16][dlo + 8];
                b1.u[0] = *(const uint4*)&lds_k[cur][32 + l16][32 + dlo];
                b1.u[1] = *(const uint4*)&lds_k[cur][32 + l16][32 + dlo + 8];
                acc2 = WMMA_BF16(qa0.v, b0.v, acc2);
                acc2 = WMMA_BF16(qa1.v, b1.v, acc2);
                b0.u[0] = *(const uint4*)&lds_k[cur][48 + l16][dlo];
                b0.u[1] = *(const uint4*)&lds_k[cur][48 + l16][dlo + 8];
                b1.u[0] = *(const uint4*)&lds_k[cur][48 + l16][32 + dlo];
                b1.u[1] = *(const uint4*)&lds_k[cur][48 + l16][32 + dlo + 8];
                acc3 = WMMA_BF16(qa0.v, b0.v, acc3);
                acc3 = WMMA_BF16(qa1.v, b1.v, acc3);
            }

            // ---- Online softmax (log2 domain), row max via 16-lane shfl_xor.
#pragma unroll
            for (int r = 0; r < 8; ++r) {
                float s0 = acc0[r] * scale2;
                float s1 = acc1[r] * scale2;
                float s2 = acc2[r] * scale2;
                float s3 = acc3[r] * scale2;
                if (s0 == padf2) s0 = -1.0e9f;   // reference quirk: score==padding masked
                if (s1 == padf2) s1 = -1.0e9f;
                if (s2 == padf2) s2 = -1.0e9f;
                if (s3 == padf2) s3 = -1.0e9f;
                if (need_mask) {                  // wave-uniform branch
                    const int row = qrow0 + r + 8 * lhalf;
                    const int c   = j0 + l16;
                    if (c > row)      s0 = -1.0e9f;
                    if (c + 16 > row) s1 = -1.0e9f;
                    if (c + 32 > row) s2 = -1.0e9f;
                    if (c + 48 > row) s3 = -1.0e9f;
                }
                float mx = fmaxf(fmaxf(s0, s1), fmaxf(s2, s3));
                mx = fmaxf(mx, __shfl_xor(mx, 1));
                mx = fmaxf(mx, __shfl_xor(mx, 2));
                mx = fmaxf(mx, __shfl_xor(mx, 4));
                mx = fmaxf(mx, __shfl_xor(mx, 8));
                const float mnew  = fmaxf(mrun[r], mx);
                const float alpha = __builtin_amdgcn_exp2f(mrun[r] - mnew);
                mrun[r] = mnew;
                const float p0 = __builtin_amdgcn_exp2f(s0 - mnew);
                const float p1 = __builtin_amdgcn_exp2f(s1 - mnew);
                const float p2 = __builtin_amdgcn_exp2f(s2 - mnew);
                const float p3 = __builtin_amdgcn_exp2f(s3 - mnew);
                lacc[r] *= alpha;
                o0[r] *= alpha; o1[r] *= alpha; o2[r] *= alpha; o3[r] *= alpha;

                const int rl = r + 8 * lhalf;
                lds_p[wave][rl][l16]      = (__bf16)p0;
                lds_p[wave][rl][l16 + 16] = (__bf16)p1;
                lds_p[wave][rl][l16 + 32] = (__bf16)p2;
                lds_p[wave][rl][l16 + 48] = (__bf16)p3;
            }

            // ---- O += P*V (8 WMMAs) and ell += P*1 (2 WMMAs).
            U16 pa0, pa1;
            pa0.u[0] = *(const uint4*)&lds_p[wave][l16][lhalf * 16];
            pa0.u[1] = *(const uint4*)&lds_p[wave][l16][8 + lhalf * 16];
            pa1.u[0] = *(const uint4*)&lds_p[wave][l16][32 + lhalf * 16];
            pa1.u[1] = *(const uint4*)&lds_p[wave][l16][40 + lhalf * 16];

            lacc = WMMA_BF16(pa0.v, onesb, lacc);
            lacc = WMMA_BF16(pa1.v, onesb, lacc);

            const int klo = lhalf * 16;
            U16 vb;
            vb.u[0] = *(const uint4*)&lds_v[cur][l16][klo];
            vb.u[1] = *(const uint4*)&lds_v[cur][l16][klo + 8];
            o0 = WMMA_BF16(pa0.v, vb.v, o0);
            vb.u[0] = *(const uint4*)&lds_v[cur][l16][klo + 32];
            vb.u[1] = *(const uint4*)&lds_v[cur][l16][klo + 40];
            o0 = WMMA_BF16(pa1.v, vb.v, o0);

            vb.u[0] = *(const uint4*)&lds_v[cur][16 + l16][klo];
            vb.u[1] = *(const uint4*)&lds_v[cur][16 + l16][klo + 8];
            o1 = WMMA_BF16(pa0.v, vb.v, o1);
            vb.u[0] = *(const uint4*)&lds_v[cur][16 + l16][klo + 32];
            vb.u[1] = *(const uint4*)&lds_v[cur][16 + l16][klo + 40];
            o1 = WMMA_BF16(pa1.v, vb.v, o1);

            vb.u[0] = *(const uint4*)&lds_v[cur][32 + l16][klo];
            vb.u[1] = *(const uint4*)&lds_v[cur][32 + l16][klo + 8];
            o2 = WMMA_BF16(pa0.v, vb.v, o2);
            vb.u[0] = *(const uint4*)&lds_v[cur][32 + l16][klo + 32];
            vb.u[1] = *(const uint4*)&lds_v[cur][32 + l16][klo + 40];
            o2 = WMMA_BF16(pa1.v, vb.v, o2);

            vb.u[0] = *(const uint4*)&lds_v[cur][48 + l16][klo];
            vb.u[1] = *(const uint4*)&lds_v[cur][48 + l16][klo + 8];
            o3 = WMMA_BF16(pa0.v, vb.v, o3);
            vb.u[0] = *(const uint4*)&lds_v[cur][48 + l16][klo + 32];
            vb.u[1] = *(const uint4*)&lds_v[cur][48 + l16][klo + 40];
            o3 = WMMA_BF16(pa1.v, vb.v, o3);
        }

#if HAVE_TDM
        if (wave == 0) __builtin_amdgcn_s_wait_tensorcnt(0);  // next tile landed
        __syncthreads();
#else
        __syncthreads();
        if (it + 1 < ntiles) {
            const int nb  = 1 - cur;
            const int row = tid >> 2;
            const int c   = (tid & 3) * 16;
            const __bf16* ks = Kp + (size_t)(it * KT + KT + row) * DD + c;
            *(uint4*)&lds_k[nb][row][c]     = *(const uint4*)(ks);
            *(uint4*)&lds_k[nb][row][c + 8] = *(const uint4*)(ks + 8);
            const __bf16* vs = Vtp + (size_t)row * SS + (it * KT + KT) + c;
            *(uint4*)&lds_v[nb][row][c]     = *(const uint4*)(vs);
            *(uint4*)&lds_v[nb][row][c + 8] = *(const uint4*)(vs + 8);
        }
        __syncthreads();
#endif
    }

    // ---- Normalize and write O (f32). Element (row r+8*lhalf, d = chunk*16 + l16).
    float* Op = out + (size_t)bh * SS * DD;
#pragma unroll
    for (int r = 0; r < 8; ++r) {
        const int row  = qrow0 + r + 8 * lhalf;
        const float rinv = __builtin_amdgcn_rcpf(lacc[r]);
        float* orow = Op + (size_t)row * DD;
        orow[l16]      = o0[r] * rinv;
        orow[16 + l16] = o1[r] * rinv;
        orow[32 + l16] = o2[r] * rinv;
        orow[48 + l16] = o3[r] * rinv;
    }
}

// ---------------------------------------------------------------------------
// Host-side launcher
// ---------------------------------------------------------------------------
extern "C" void kernel_launch(void* const* d_in, const int* in_sizes, int n_in,
                              void* d_out, int out_size, void* d_ws, size_t ws_size,
                              hipStream_t stream) {
    const float* q = (const float*)d_in[0];
    const float* k = (const float*)d_in[1];
    const float* v = (const float*)d_in[2];
    const int* is_masked = (const int*)d_in[3];
    const int* padding   = (const int*)d_in[4];
    float* out = (float*)d_out;

    const size_t NE = (size_t)BB * HH * SS * DD;   // 4,194,304 elems per tensor
    __bf16* wq  = (__bf16*)d_ws;
    __bf16* wk  = wq + NE;
    __bf16* wvt = wk + NE;                         // V^T: [bh][d][s]

    const unsigned cvt_blocks = (unsigned)(NE / (256u * 4u));
    cvt_f32_to_bf16<<<cvt_blocks, 256, 0, stream>>>(q, wq, (unsigned)NE);
    cvt_f32_to_bf16<<<cvt_blocks, 256, 0, stream>>>(k, wk, (unsigned)NE);
    dim3 tgrid(SS / 64, BHN);
    cvt_transpose_v<<<tgrid, 256, 0, stream>>>(v, wvt);

    dim3 grid(SS / QTILE, BHN);
    fa_fwd_kernel<<<grid, 256, 0, stream>>>(wq, wk, wvt, is_masked, padding, out);
}
